// MultiHeadAttention_55121610277393
// MI455X (gfx1250) — matrix operand order repaired and checked by reading, not compiled, not run on hardware
//
#include <hip/hip_runtime.h>
#include <hip/hip_bf16.h>

// ---------------------------------------------------------------------------
// Multi-head attention for MI455X (gfx1250), wave32 + v_wmma_f32_16x16x32_f16
// GEMM waves produce 16x64 tiles (one A load feeds 4 wmmas) with explicit
// double-buffering so next-step loads overlap current-step wmmas.
// ---------------------------------------------------------------------------

typedef __attribute__((ext_vector_type(16))) _Float16 v16h;
typedef __attribute__((ext_vector_type(8)))  _Float16 v8h;
typedef __attribute__((ext_vector_type(8)))  float    v8f;

#define S_LEN 2048
#define D_DIM 1024
#define H_NUM 16
#define DKV   64

// Load a WMMA 16-half operand as two contiguous 8-half (16B) chunks.
static __device__ __forceinline__ v16h ld16(const _Float16* p0, const _Float16* p1) {
  v16h r;
  ((v8h*)&r)[0] = *(const v8h*)p0;
  ((v8h*)&r)[1] = *(const v8h*)p1;
  return r;
}

static __device__ __forceinline__ v8f wmma16(v16h a, v16h b, v8f c) {
  // (neg_a, A, neg_b, B, c_mod, C, reuse_a, reuse_b)
  return __builtin_amdgcn_wmma_f32_16x16x32_f16(false, a, false, b, (short)0, c,
                                                false, false);
}

// ---------------------------------------------------------------------------
// Stage 0: precision conversion / transposition into f16 staging buffers
// ---------------------------------------------------------------------------
__global__ void cvt_f16_kernel(const float* __restrict__ in,
                               _Float16* __restrict__ out, int n) {
  int i = blockIdx.x * blockDim.x + threadIdx.x;
  if (i < n) out[i] = (_Float16)in[i];
}

// in: [batch][rows][cols] f32  ->  out: [batch][cols][rows] f16
__global__ void transpose_f16_kernel(const float* __restrict__ in,
                                     _Float16* __restrict__ out,
                                     int batch, int rows, int cols) {
  int i = blockIdx.x * blockDim.x + threadIdx.x;
  int total = batch * rows * cols;
  if (i >= total) return;
  int b  = i / (rows * cols);
  int rc = i % (rows * cols);
  int r  = rc / cols;
  int c  = rc % cols;
  out[(size_t)b * rows * cols + (size_t)c * rows + r] = (_Float16)in[i];
}

// ---------------------------------------------------------------------------
// Stage 1: per-head projections  P[h] = E (S x D) * W[h] (D x 64)
//   E:  [S][D] f16 row-major (A operand)
//   Wt: [H][64][D] f16 (pre-transposed: B operand column n is contiguous)
// One wave computes a full 16x64 tile; double-buffered K loop.
//   transposeOut==0 -> P[h][s][c] f16 ; ==1 -> P[h][c][s] f16 (for V)
//   `scale` folds 1/sqrt(dk) into Q.
// ---------------------------------------------------------------------------
__global__ __launch_bounds__(128) void proj_kernel(const _Float16* __restrict__ E,
                                                   const _Float16* __restrict__ Wt,
                                                   _Float16* __restrict__ Pout,
                                                   float scale, int transposeOut) {
  const int lane = threadIdx.x & 31;
  const int wave = threadIdx.x >> 5;
  const int rt   = blockIdx.x * 4 + wave;      // 0..127 row tiles
  const int h    = blockIdx.y;
  const int row0 = rt << 4;
  const int n    = lane & 15;
  const int g    = lane >> 4;
  const int kbA  = g ? 16 : 0;    // A layout: lane<16 holds k 0..15
  const int kbB  = g ? 16 : 0;    // B layout: lane<16 holds k 0..15

  const _Float16* arow = E + (size_t)(row0 + n) * D_DIM + kbA;
  const _Float16* b0 = Wt + ((size_t)h * DKV + 0  + n) * D_DIM + kbB;
  const _Float16* b1 = Wt + ((size_t)h * DKV + 16 + n) * D_DIM + kbB;
  const _Float16* b2 = Wt + ((size_t)h * DKV + 32 + n) * D_DIM + kbB;
  const _Float16* b3 = Wt + ((size_t)h * DKV + 48 + n) * D_DIM + kbB;

  v8f c0 = {}, c1 = {}, c2 = {}, c3 = {};
  // Prologue: load k-step 0.
  v16h a  = ld16(arow, arow + 8);
  v16h B0 = ld16(b0, b0 + 8);
  v16h B1 = ld16(b1, b1 + 8);
  v16h B2 = ld16(b2, b2 + 8);
  v16h B3 = ld16(b3, b3 + 8);
  for (int k0 = 0; k0 < D_DIM; k0 += 32) {
    const int kn = (k0 + 32 < D_DIM) ? k0 + 32 : 0;   // wrap: dead but in-bounds
    // Issue next-step loads before consuming current registers.
    v16h an  = ld16(arow + kn, arow + kn + 8);
    v16h B0n = ld16(b0 + kn, b0 + kn + 8);
    v16h B1n = ld16(b1 + kn, b1 + kn + 8);
    v16h B2n = ld16(b2 + kn, b2 + kn + 8);
    v16h B3n = ld16(b3 + kn, b3 + kn + 8);
    c0 = wmma16(a, B0, c0);
    c1 = wmma16(a, B1, c1);
    c2 = wmma16(a, B2, c2);
    c3 = wmma16(a, B3, c3);
    a = an; B0 = B0n; B1 = B1n; B2 = B2n; B3 = B3n;
  }

#pragma unroll
  for (int r = 0; r < 8; ++r) {
    const int m = r + 8 * g;                  // C layout: VGPR r -> row r (+8 hi half)
    const _Float16 v0 = (_Float16)(c0[r] * scale);
    const _Float16 v1 = (_Float16)(c1[r] * scale);
    const _Float16 v2 = (_Float16)(c2[r] * scale);
    const _Float16 v3 = (_Float16)(c3[r] * scale);
    if (!transposeOut) {
      _Float16* dst = Pout + ((size_t)h * S_LEN + row0 + m) * DKV + n;
      dst[0] = v0; dst[16] = v1; dst[32] = v2; dst[48] = v3;
    } else {
      _Float16* dst = Pout + ((size_t)h * DKV + n) * S_LEN + row0 + m;
      dst[0 * 16 * S_LEN] = v0;
      dst[1 * 16 * S_LEN] = v1;
      dst[2 * 16 * S_LEN] = v2;
      dst[3 * 16 * S_LEN] = v3;
    }
  }
}

// ---------------------------------------------------------------------------
// Stage 2: flash attention. One wave owns (head h, 16 query rows).
//   Q:  [H][S][64] f16 (pre-scaled by 1/8)
//   K:  [H][S][64] f16
//   Vt: [H][64][S] f16 (transposed so PV B-operand is contiguous over t)
//   ctx:[S][H*64]  f16 (concat layout for the output projection)
// Streams keys 32 at a time: 4 QK wmmas -> online softmax (shfl_xor row
// reductions within 16-lane groups) -> P via LDS (D-layout -> A-layout
// transpose) -> 4 PV wmmas.
// ---------------------------------------------------------------------------
__global__ __launch_bounds__(128) void attn_kernel(const _Float16* __restrict__ Q,
                                                   const _Float16* __restrict__ K,
                                                   const _Float16* __restrict__ Vt,
                                                   _Float16* __restrict__ ctx) {
  __shared__ _Float16 ldsP[4][16 * 32];        // 1 KB per wave
  const int lane = threadIdx.x & 31;
  const int wave = threadIdx.x >> 5;
  const int tile = blockIdx.x * 4 + wave;      // 0..2047 = H * (S/16)
  const int h    = tile >> 7;
  const int q0   = (tile & 127) << 4;
  const int n    = lane & 15;
  const int g    = lane >> 4;
  const int kbA  = g ? 16 : 0;
  const int kbB  = g ? 16 : 0;

  const _Float16* Qh = Q  + (size_t)h * S_LEN * DKV;
  const _Float16* Kh = K  + (size_t)h * S_LEN * DKV;
  const _Float16* Vh = Vt + (size_t)h * DKV * S_LEN;
  _Float16* P = ldsP[wave];

  // Q rows for this tile, held in A-operand registers for the whole stream.
  const _Float16* qrow = Qh + (size_t)(q0 + n) * DKV;
  const v16h aq0 = ld16(qrow + kbA,      qrow + kbA + 8);        // k 0..31
  const v16h aq1 = ld16(qrow + 32 + kbA, qrow + 32 + kbA + 8);   // k 32..63

  v8f o0 = {}, o1 = {}, o2 = {}, o3 = {};
  float mr[8], lr[8];
#pragma unroll
  for (int r = 0; r < 8; ++r) { mr[r] = -3.0e38f; lr[r] = 0.0f; }

  for (int t0 = 0; t0 < S_LEN; t0 += 32) {
    // ---- scores for two 16-key tiles: S = Q * K^T (K reduction = 64) ----
    const _Float16* ka = Kh + (size_t)(t0 + n) * DKV;        // keys t0..t0+15
    const _Float16* kb = Kh + (size_t)(t0 + 16 + n) * DKV;   // keys t0+16..t0+31
    v8f sa = {}, sb = {};
    sa = wmma16(aq0, ld16(ka + kbB,      ka + kbB + 8),      sa);
    sa = wmma16(aq1, ld16(ka + 32 + kbB, ka + 32 + kbB + 8), sa);
    sb = wmma16(aq0, ld16(kb + kbB,      kb + kbB + 8),      sb);
    sb = wmma16(aq1, ld16(kb + 32 + kbB, kb + 32 + kbB + 8), sb);

    // ---- online softmax over the 32 new keys ----
    float alpha[8];
#pragma unroll
    for (int r = 0; r < 8; ++r) {
      float x = fmaxf(sa[r], sb[r]);
      x = fmaxf(x, __shfl_xor(x, 1));
      x = fmaxf(x, __shfl_xor(x, 2));
      x = fmaxf(x, __shfl_xor(x, 4));
      x = fmaxf(x, __shfl_xor(x, 8));          // row max within 16-lane group
      const float nm = fmaxf(mr[r], x);
      alpha[r] = __expf(mr[r] - nm);
      const float pa = __expf(sa[r] - nm);
      const float pb = __expf(sb[r] - nm);
      float rs = pa + pb;
      rs += __shfl_xor(rs, 1);
      rs += __shfl_xor(rs, 2);
      rs += __shfl_xor(rs, 4);
      rs += __shfl_xor(rs, 8);                 // row sum
      lr[r] = lr[r] * alpha[r] + rs;
      mr[r] = nm;
      const int qr = r + 8 * g;                // C-layout row index
      P[qr * 32 + n]      = (_Float16)pa;      // D-layout -> LDS [16 q][32 t]
      P[qr * 32 + 16 + n] = (_Float16)pb;
    }
#pragma unroll
    for (int r = 0; r < 8; ++r) {
      o0[r] *= alpha[r]; o1[r] *= alpha[r]; o2[r] *= alpha[r]; o3[r] *= alpha[r];
    }

    // LDS is in-order per wave; explicit wait keeps the RAW airtight.
    asm volatile("s_wait_dscnt 0x0" ::: "memory");

    // Reload probabilities in A-operand layout (rows = queries, k = 32 keys).
    const v16h ap = ld16(P + n * 32 + kbA, P + n * 32 + kbA + 8);

    // ---- O += P * V  (4 column tiles of 16, one shared A operand) ----
    const _Float16* v0 = Vh + (size_t)(0  + n) * S_LEN + t0 + kbB;
    const _Float16* v1 = Vh + (size_t)(16 + n) * S_LEN + t0 + kbB;
    const _Float16* v2 = Vh + (size_t)(32 + n) * S_LEN + t0 + kbB;
    const _Float16* v3 = Vh + (size_t)(48 + n) * S_LEN + t0 + kbB;
    o0 = wmma16(ap, ld16(v0, v0 + 8), o0);
    o1 = wmma16(ap, ld16(v1, v1 + 8), o1);
    o2 = wmma16(ap, ld16(v2, v2 + 8), o2);
    o3 = wmma16(ap, ld16(v3, v3 + 8), o3);
  }

  // ---- normalize and emit ctx in concat layout [s][h*64 + v] ----
#pragma unroll
  for (int r = 0; r < 8; ++r) {
    const size_t s = (size_t)(q0 + r + 8 * g);
    const float inv = 1.0f / lr[r];
    _Float16* dst = ctx + s * (H_NUM * DKV) + h * DKV + n;
    dst[0]  = (_Float16)(o0[r] * inv);
    dst[16] = (_Float16)(o1[r] * inv);
    dst[32] = (_Float16)(o2[r] * inv);
    dst[48] = (_Float16)(o3[r] * inv);
  }
}

// ---------------------------------------------------------------------------
// Stage 3: output projection  out = ctx (S x 1024) * W_out (1024 x 1024), f32 out
//   Bt: [1024 n][1024 k] f16 (W_out pre-transposed)
// One wave computes a 16x64 tile; double-buffered K loop.
// ---------------------------------------------------------------------------
__global__ __launch_bounds__(128) void outproj_kernel(const _Float16* __restrict__ A,
                                                      const _Float16* __restrict__ Bt,
                                                      float* __restrict__ out) {
  const int lane = threadIdx.x & 31;
  const int wave = threadIdx.x >> 5;
  const int tile = blockIdx.x * 4 + wave;      // 0..2047 (128 row x 16 col blocks)
  const int row0 = (tile >> 4) << 4;
  const int col0 = (tile & 15) << 6;           // 64-wide column block
  const int n    = lane & 15;
  const int g    = lane >> 4;
  const int kbA  = g ? 16 : 0;
  const int kbB  = g ? 16 : 0;

  const _Float16* arow = A + (size_t)(row0 + n) * D_DIM + kbA;
  const _Float16* b0 = Bt + (size_t)(col0 + 0  + n) * D_DIM + kbB;
  const _Float16* b1 = Bt + (size_t)(col0 + 16 + n) * D_DIM + kbB;
  const _Float16* b2 = Bt + (size_t)(col0 + 32 + n) * D_DIM + kbB;
  const _Float16* b3 = Bt + (size_t)(col0 + 48 + n) * D_DIM + kbB;

  v8f c0 = {}, c1 = {}, c2 = {}, c3 = {};
  v16h a  = ld16(arow, arow + 8);
  v16h B0 = ld16(b0, b0 + 8);
  v16h B1 = ld16(b1, b1 + 8);
  v16h B2 = ld16(b2, b2 + 8);
  v16h B3 = ld16(b3, b3 + 8);
  for (int k0 = 0; k0 < D_DIM; k0 += 32) {
    const int kn = (k0 + 32 < D_DIM) ? k0 + 32 : 0;   // wrap: dead but in-bounds
    v16h an  = ld16(arow + kn, arow + kn + 8);
    v16h B0n = ld16(b0 + kn, b0 + kn + 8);
    v16h B1n = ld16(b1 + kn, b1 + kn + 8);
    v16h B2n = ld16(b2 + kn, b2 + kn + 8);
    v16h B3n = ld16(b3 + kn, b3 + kn + 8);
    c0 = wmma16(a, B0, c0);
    c1 = wmma16(a, B1, c1);
    c2 = wmma16(a, B2, c2);
    c3 = wmma16(a, B3, c3);
    a = an; B0 = B0n; B1 = B1n; B2 = B2n; B3 = B3n;
  }
#pragma unroll
  for (int r = 0; r < 8; ++r) {
    float* dst = out + (size_t)(row0 + r + 8 * g) * D_DIM + col0 + n;
    dst[0] = c0[r]; dst[16] = c1[r]; dst[32] = c2[r]; dst[48] = c3[r];
  }
}

// ---------------------------------------------------------------------------
// Host-side orchestration (graph-capture safe: only async launches on stream)
// ---------------------------------------------------------------------------
extern "C" void kernel_launch(void* const* d_in, const int* in_sizes, int n_in,
                              void* d_out, int out_size, void* d_ws, size_t ws_size,
                              hipStream_t stream) {
  const float* Eq = (const float*)d_in[0];
  const float* Ek = (const float*)d_in[1];
  const float* Ev = (const float*)d_in[2];
  const float* Wq = (const float*)d_in[3];
  const float* Wk = (const float*)d_in[4];
  const float* Wv = (const float*)d_in[5];
  const float* Wo = (const float*)d_in[6];
  float* out = (float*)d_out;

  // f16 staging buffers carved out of d_ws (~37.7 MB total).
  _Float16* ws = (_Float16*)d_ws;
  size_t off = 0;
  _Float16* eq16 = ws + off; off += (size_t)S_LEN * D_DIM;
  _Float16* ek16 = ws + off; off += (size_t)S_LEN * D_DIM;
  _Float16* ev16 = ws + off; off += (size_t)S_LEN * D_DIM;
  _Float16* wqt  = ws + off; off += (size_t)H_NUM * DKV * D_DIM;
  _Float16* wkt  = ws + off; off += (size_t)H_NUM * DKV * D_DIM;
  _Float16* wvt  = ws + off; off += (size_t)H_NUM * DKV * D_DIM;
  _Float16* wot  = ws + off; off += (size_t)D_DIM * D_DIM;
  _Float16* Qb   = ws + off; off += (size_t)H_NUM * S_LEN * DKV;
  _Float16* Kb   = ws + off; off += (size_t)H_NUM * S_LEN * DKV;
  _Float16* Vtb  = ws + off; off += (size_t)H_NUM * S_LEN * DKV;
  _Float16* ctx  = ws + off; off += (size_t)S_LEN * H_NUM * DKV;
  (void)ws_size; (void)in_sizes; (void)n_in; (void)out_size;

  const int nE = S_LEN * D_DIM;                         // 2 097 152
  cvt_f16_kernel<<<nE / 256, 256, 0, stream>>>(Eq, eq16, nE);
  cvt_f16_kernel<<<nE / 256, 256, 0, stream>>>(Ek, ek16, nE);
  cvt_f16_kernel<<<nE / 256, 256, 0, stream>>>(Ev, ev16, nE);

  const int nW = H_NUM * D_DIM * DKV;                   // 1 048 576
  transpose_f16_kernel<<<nW / 256, 256, 0, stream>>>(Wq, wqt, H_NUM, D_DIM, DKV);
  transpose_f16_kernel<<<nW / 256, 256, 0, stream>>>(Wk, wkt, H_NUM, D_DIM, DKV);
  transpose_f16_kernel<<<nW / 256, 256, 0, stream>>>(Wv, wvt, H_NUM, D_DIM, DKV);
  transpose_f16_kernel<<<(D_DIM * D_DIM) / 256, 256, 0, stream>>>(Wo, wot, 1, D_DIM, D_DIM);

  // Projections: one wave per 16-row tile per head (16x64 output per wave);
  // fold 1/sqrt(dk)=0.125 into Q; store V transposed for the PV B-operand.
  dim3 pb(128), pg(32, H_NUM);
  proj_kernel<<<pg, pb, 0, stream>>>(eq16, wqt, Qb, 0.125f, 0);
  proj_kernel<<<pg, pb, 0, stream>>>(ek16, wkt, Kb, 1.0f, 0);
  proj_kernel<<<pg, pb, 0, stream>>>(ev16, wvt, Vtb, 1.0f, 1);

  // Flash attention: 2048 (head, qtile) waves / 4 per block.
  attn_kernel<<<512, 128, 0, stream>>>(Qb, Kb, Vtb, ctx);

  // Output projection: 2048 16x64 tiles / 4 waves per block.
  outproj_kernel<<<512, 128, 0, stream>>>(ctx, wot, out);
}
